// LSTMTagger_65008624992512
// MI455X (gfx1250) — matrix operand order repaired and checked by reading, compiled, not run on hardware
//
#include <hip/hip_runtime.h>
#include <hip/hip_bf16.h>
#include <math.h>

// ---------------------------------------------------------------------------
// Problem sizes (fixed by the reference)
// ---------------------------------------------------------------------------
#define TT    4096
#define EMB   512
#define HID   512
#define G4    2048      // 4*HID
#define TAGS  64

typedef __attribute__((ext_vector_type(16))) __bf16 v16bf;
typedef __attribute__((ext_vector_type(8)))  __bf16 v8bf;
typedef __attribute__((ext_vector_type(8)))  float  v8f;

// ---------------------------------------------------------------------------
// Small prep kernels
// ---------------------------------------------------------------------------
__global__ void k_f32_to_bf16(const float* __restrict__ in,
                              __bf16* __restrict__ out, int n) {
    int i = blockIdx.x * blockDim.x + threadIdx.x;
    if (i < n) out[i] = (__bf16)in[i];
}

__global__ void k_add_bias(const float* __restrict__ a,
                           const float* __restrict__ b,
                           float* __restrict__ out, int n) {
    int i = blockIdx.x * blockDim.x + threadIdx.x;
    if (i < n) out[i] = a[i] + b[i];
}

// x_bf16[t, e] = (bf16) emb[sentence[t], e]
__global__ void k_gather_bf16(const int* __restrict__ sent,
                              const float* __restrict__ emb,
                              __bf16* __restrict__ out) {
    int i = blockIdx.x * blockDim.x + threadIdx.x;   // over TT*EMB
    if (i >= TT * EMB) return;
    int t = i / EMB, e = i - t * EMB;
    out[i] = (__bf16)emb[(size_t)sent[t] * EMB + e];
}

// ---------------------------------------------------------------------------
// WMMA GEMM:  C[M,N] = A[M,K](bf16) * B[N,K](bf16)^T + bias[N]
// One wave computes one 16x16 C tile; K stepped by 32 with
// v_wmma_f32_16x16x32_bf16.  Fragment layouts per CDNA5 ISA 7.12.2:
//   A (16x32 bf16): lane L holds row M=L%16; half=L/16 selects K-subblocks
//       {half*8 + 0..7} and {16 + half*8 + 0..7}  -> two contiguous 16B loads
//   B (32x16 bf16): lane L holds column N=L%16, same K-subblocks as A
//       {half*8 + 0..7} and {16 + half*8 + 0..7} of a W row (C = A * B^T)
//   C/D (16x16 f32): lane L -> n = L%16, vgpr r -> m = r + 8*(L/16)
// ---------------------------------------------------------------------------
__global__ void k_gemm_bf16_wmma(const __bf16* __restrict__ A,   // [M,K]
                                 const __bf16* __restrict__ B,   // [N,K]
                                 const float*  __restrict__ bias,// [N]
                                 float* __restrict__ C,          // [M,N]
                                 int M, int N, int K) {
    const int lane = threadIdx.x & 31;
    const int wave = threadIdx.x >> 5;
    const int tile = blockIdx.x * (blockDim.x >> 5) + wave;
    const int tilesN = N >> 4;
    const int tm = tile / tilesN;
    const int tn = tile - tm * tilesN;
    if (tm * 16 >= M) return;                 // uniform per wave

    const int half = lane >> 4;
    const int l16  = lane & 15;
    const __bf16* Arow = A + (size_t)(tm * 16 + l16) * K;
    const __bf16* Brow = B + (size_t)(tn * 16 + l16) * K;

    v8f c = {};
    for (int k0 = 0; k0 < K; k0 += 32) {
        v8bf a0 = *(const v8bf*)(Arow + k0 + half * 8);
        v8bf a1 = *(const v8bf*)(Arow + k0 + 16 + half * 8);
        v8bf b0 = *(const v8bf*)(Brow + k0 + half * 8);
        v8bf b1 = *(const v8bf*)(Brow + k0 + 16 + half * 8);
        v16bf a, b;
#pragma unroll
        for (int i = 0; i < 8; ++i) {
            a[i] = a0[i]; a[8 + i] = a1[i];
            b[i] = b0[i]; b[8 + i] = b1[i];
        }
        c = __builtin_amdgcn_wmma_f32_16x16x32_bf16(
                /*neg_a=*/false, a, /*neg_b=*/false, b,
                /*c_mod=*/(short)0, c, /*reuse_a=*/false, /*reuse_b=*/false);
    }

    const float bv = bias ? bias[tn * 16 + l16] : 0.0f;
#pragma unroll
    for (int r = 0; r < 8; ++r) {
        int m = tm * 16 + r + half * 8;
        C[(size_t)m * N + tn * 16 + l16] = c[r] + bv;
    }
}

// ---------------------------------------------------------------------------
// Sequential LSTM scan.
// 128 persistent workgroups; WG wg owns hidden indices k in [4*wg, 4*wg+4)
// -> 16 rows of W_hh (4 per gate), transposed + padded into LDS (stride 17
// floats so a full wave hits 32 distinct banks). Cell state c lives in
// registers of threads 0..3. Cross-WG handoff of h via hs[] in global memory
// with a per-step release/acquire counter (reset by hipMemsetAsync per call).
// ---------------------------------------------------------------------------
#define SCAN_WGS 128
#define SCAN_THR 256
#define SCAN_HK  4      // hidden indices per WG
#define SCAN_RWS 16     // 4 gates * SCAN_HK rows of W_hh
#define WPAD     17     // LDS row stride (floats) to kill bank conflicts

__device__ __forceinline__ float sigmoidf_(float x) {
    return 1.0f / (1.0f + __expf(-x));
}

__global__ void k_lstm_scan(const float* __restrict__ xg,   // [TT, G4]
                            const float* __restrict__ Whh,  // [G4, HID]
                            float* __restrict__ hs,         // [TT, HID]
                            unsigned* __restrict__ cnt) {   // [TT], zeroed
    extern __shared__ float sm[];
    float* w       = sm;                      // [HID][SCAN_RWS] stride WPAD
    float* hbuf    = w + HID * WPAD;          // [HID]
    float* partial = hbuf + HID;              // [16 parts][SCAN_RWS]
    float* gv      = partial + 16 * SCAN_RWS; // [SCAN_RWS]

    const int tid   = threadIdx.x;
    const int kbase = blockIdx.x * SCAN_HK;

    // Stage this WG's 16 W_hh rows, transposed: w[j*WPAD + lr] = Whh[grow(lr), j]
    for (int idx = tid; idx < SCAN_RWS * HID; idx += SCAN_THR) {
        int lr = idx & (SCAN_RWS - 1);
        int j  = idx >> 4;
        int grow = (lr >> 2) * HID + kbase + (lr & 3);
        w[j * WPAD + lr] = Whh[(size_t)grow * HID + j];
    }
    __syncthreads();

    const int row  = tid & 15;   // which of our 16 gate rows
    const int part = tid >> 4;   // 0..15 : 32-wide slice of h
    float c_reg = 0.0f;

    for (int s = 0; s < TT; ++s) {
        float acc = 0.0f;
        if (s > 0) {
            // stage h_{s-1}
            hbuf[tid]       = hs[(size_t)(s - 1) * HID + tid];
            hbuf[tid + 256] = hs[(size_t)(s - 1) * HID + tid + 256];
            __syncthreads();
            const int jb = part * 32;
#pragma unroll 8
            for (int j = 0; j < 32; ++j)
                acc += hbuf[jb + j] * w[(jb + j) * WPAD + row];
        }
        partial[part * SCAN_RWS + row] = acc;
        __syncthreads();

        if (tid < SCAN_RWS) {
            int grow = (tid >> 2) * HID + kbase + (tid & 3);
            float sum = xg[(size_t)s * G4 + grow];
#pragma unroll
            for (int p = 0; p < 16; ++p) sum += partial[p * SCAN_RWS + tid];
            gv[tid] = sum;
        }
        __syncthreads();

        if (tid < SCAN_HK) {                  // PyTorch gate order i,f,g,o
            float ig = sigmoidf_(gv[tid]);
            float fg = sigmoidf_(gv[SCAN_HK + tid]);
            float gg = tanhf(gv[2 * SCAN_HK + tid]);
            float og = sigmoidf_(gv[3 * SCAN_HK + tid]);
            c_reg = fg * c_reg + ig * gg;
            hs[(size_t)s * HID + kbase + tid] = og * tanhf(c_reg);
        }
        __threadfence();
        __syncthreads();
        if (tid == 0)
            __atomic_fetch_add(&cnt[s], 1u, __ATOMIC_ACQ_REL);
        if (s + 1 < TT) {
            if (tid == 0) {
                while (__atomic_load_n(&cnt[s], __ATOMIC_ACQUIRE) < SCAN_WGS)
                    __builtin_amdgcn_s_sleep(1);
            }
            __syncthreads();
        }
    }
}

// ---------------------------------------------------------------------------
// In-place log_softmax over 64-wide rows: one wave per row (2 elems/lane).
// ---------------------------------------------------------------------------
__global__ void k_log_softmax64(float* __restrict__ x, int rows) {
    int lane = threadIdx.x & 31;
    int wave = threadIdx.x >> 5;
    int r = blockIdx.x * (blockDim.x >> 5) + wave;
    if (r >= rows) return;
    float* p = x + (size_t)r * TAGS;
    float x0 = p[lane], x1 = p[lane + 32];
    float m = fmaxf(x0, x1);
#pragma unroll
    for (int off = 16; off; off >>= 1) m = fmaxf(m, __shfl_xor(m, off, 32));
    float s = __expf(x0 - m) + __expf(x1 - m);
#pragma unroll
    for (int off = 16; off; off >>= 1) s += __shfl_xor(s, off, 32);
    float ls = m + __logf(s);
    p[lane]      = x0 - ls;
    p[lane + 32] = x1 - ls;
}

// ---------------------------------------------------------------------------
// Launch
// ---------------------------------------------------------------------------
extern "C" void kernel_launch(void* const* d_in, const int* in_sizes, int n_in,
                              void* d_out, int out_size, void* d_ws, size_t ws_size,
                              hipStream_t stream) {
    const int*   sent  = (const int*)  d_in[0];
    const float* emb   = (const float*)d_in[1];
    const float* W_ih  = (const float*)d_in[2];
    const float* W_hh  = (const float*)d_in[3];
    const float* b_ih  = (const float*)d_in[4];
    const float* b_hh  = (const float*)d_in[5];
    const float* W_out = (const float*)d_in[6];
    const float* b_out = (const float*)d_in[7];
    float* out = (float*)d_out;                      // [TT, TAGS]

    // ---- workspace layout (~52.5 MB) ----
    char* ws = (char*)d_ws;
    float* xg      = (float*)ws;  ws += (size_t)TT * G4  * sizeof(float);   // 32 MB
    float* hs      = (float*)ws;  ws += (size_t)TT * HID * sizeof(float);   //  8 MB
    __bf16* xbf    = (__bf16*)ws; ws += (size_t)TT * EMB * sizeof(__bf16);  //  4 MB
    __bf16* wihbf  = (__bf16*)ws; ws += (size_t)G4 * EMB * sizeof(__bf16);  //  2 MB
    __bf16* hsbf   = (__bf16*)ws; ws += (size_t)TT * HID * sizeof(__bf16);  //  4 MB
    __bf16* woutbf = (__bf16*)ws; ws += (size_t)TAGS * HID * sizeof(__bf16);
    float* bias    = (float*)ws;  ws += (size_t)G4 * sizeof(float);
    unsigned* cnt  = (unsigned*)ws;

    // ---- 1) prep: bf16 weight copies, fused bias, embedding gather ----
    k_f32_to_bf16<<<(G4 * EMB + 255) / 256, 256, 0, stream>>>(W_ih, wihbf, G4 * EMB);
    k_f32_to_bf16<<<(TAGS * HID + 255) / 256, 256, 0, stream>>>(W_out, woutbf, TAGS * HID);
    k_add_bias   <<<(G4 + 255) / 256, 256, 0, stream>>>(b_ih, b_hh, bias, G4);
    k_gather_bf16<<<(TT * EMB + 255) / 256, 256, 0, stream>>>(sent, emb, xbf);

    // ---- 2) x_gates = x @ W_ih^T + bias : WMMA bf16 GEMM (4096x2048x512) ----
    // tiles = 256 * 128 = 32768 ; 8 waves / block -> 4096 blocks
    k_gemm_bf16_wmma<<<4096, 256, 0, stream>>>(xbf, wihbf, bias, xg, TT, G4, EMB);

    // ---- 3) sequential scan : 128 persistent WGs, per-step counter sync ----
    hipMemsetAsync(cnt, 0, (size_t)TT * sizeof(unsigned), stream);
    size_t smem = (size_t)(HID * WPAD + HID + 16 * SCAN_RWS + SCAN_RWS) * sizeof(float);
    k_lstm_scan<<<SCAN_WGS, SCAN_THR, smem, stream>>>(xg, W_hh, hs, cnt);

    // ---- 4) tag logits = hs @ W_out^T + b_out : WMMA GEMM (4096x64x512) ----
    k_f32_to_bf16<<<(TT * HID + 255) / 256, 256, 0, stream>>>(hs, hsbf, TT * HID);
    // tiles = 256 * 4 = 1024 ; 8 waves / block -> 128 blocks
    k_gemm_bf16_wmma<<<128, 256, 0, stream>>>(hsbf, woutbf, b_out, out, TT, TAGS, HID);

    // ---- 5) in-place log_softmax over the 64 tags ----
    k_log_softmax64<<<512, 256, 0, stream>>>(out, TT);
}
